// ActorCritic_73160472920261
// MI455X (gfx1250) — compile-verified, matrix operand order checked by reading, not run on hardware
//
#include <hip/hip_runtime.h>
#include <math.h>

typedef __bf16 bf16;
typedef __attribute__((ext_vector_type(16))) __bf16 v16bf;
typedef __attribute__((ext_vector_type(8)))  float  v8f;

// ---------------------------------------------------------------- utilities
__device__ __forceinline__ float u01_hash(unsigned x) {
  x ^= x >> 17; x *= 0xed5ad4bbu;
  x ^= x >> 11; x *= 0xac4c1b51u;
  x ^= x >> 15; x *= 0x31848babu;
  x ^= x >> 14;
  float u = (float)(x >> 8) * (1.0f / 16777216.0f);
  return fminf(fmaxf(u, 1e-6f), 1.0f - 1e-6f);
}

// ------------------------------------------------- edge message + scatter-add
// msg = relu(h[src] + edge_attr @ We + be) ; agg[dst] += msg   (d = 128)
__global__ __launch_bounds__(256) void edge_msg128_kernel(
    const float* __restrict__ h, const int* __restrict__ src,
    const int* __restrict__ dst, const float* __restrict__ ea,
    const float* __restrict__ We, const float* __restrict__ be,
    float* __restrict__ agg, int E) {
  int gt = blockIdx.x * 256 + threadIdx.x;
  int e = gt >> 7, c = gt & 127;
  if (e >= E) return;
  int s = src[e], d = dst[e];
  const float* a4 = ea + e * 4;
  float a0 = a4[0], a1 = a4[1], a2 = a4[2], a3 = a4[3];
  float m = h[s * 128 + c]
          + a0 * We[c] + a1 * We[128 + c] + a2 * We[256 + c] + a3 * We[384 + c]
          + be[c];
  m = fmaxf(m, 0.0f);
  atomicAdd(&agg[d * 128 + c], m);
}

// same, d = 8 (layer 0: h == x, We is [4,8])
__global__ __launch_bounds__(256) void edge_msg8_kernel(
    const float* __restrict__ h, const int* __restrict__ src,
    const int* __restrict__ dst, const float* __restrict__ ea,
    const float* __restrict__ We, const float* __restrict__ be,
    float* __restrict__ agg, int E) {
  int gt = blockIdx.x * 256 + threadIdx.x;
  int e = gt >> 3, c = gt & 7;
  if (e >= E) return;
  int s = src[e], d = dst[e];
  const float* a4 = ea + e * 4;
  float m = h[s * 8 + c]
          + a4[0] * We[c] + a4[1] * We[8 + c] + a4[2] * We[16 + c] + a4[3] * We[24 + c]
          + be[c];
  m = fmaxf(m, 0.0f);
  atomicAdd(&agg[d * 8 + c], m);
}

// ------------------------------------------------------------ WMMA dense GEMM
// Out[N,128] = act((X[N,K] (+Agg[N,K])) @ W[K,128] + bias)
// Block: 128 threads = 4 waves, 64 rows/block; wave w -> rows [w*16, w*16+16).
// A/B staged in LDS pre-swizzled to the 16-bit WMMA fragment layout so each
// lane's fragment is one contiguous 32-byte LDS read.
__global__ __launch_bounds__(128) void dense_wmma_kernel(
    const float* __restrict__ X, const float* __restrict__ Agg,
    const float* __restrict__ W, const float* __restrict__ bias,
    float* __restrict__ Out, int K, int applyRelu) {
  const int Kpad = (K + 31) & ~31;
  const int nkb = Kpad >> 5;                 // # of K=32 steps (1 or 4 here)
  __shared__ __align__(32) bf16 As[64 * 4 * 32];   // 64 rows * nkb * 2 * v16
  __shared__ __align__(32) bf16 Bs[8 * 4 * 32 * 16];
  const int tid = threadIdx.x;
  const int r0 = blockIdx.x * 64;

  // Stage A = X(+Agg), swizzled: As[((row*nkb+kkb)*2+hi)*16 + e]
  for (int idx = tid; idx < 64 * nkb * 32; idx += 128) {
    int e = idx & 15;
    int t = idx >> 4;
    int hi = t & 1; t >>= 1;
    int kkb = t % nkb, row = t / nkb;
    int col = kkb * 32 + e + (e & 8) + hi * 8;   // fragment K index
    float v = 0.0f;
    if (col < K) {
      int g = (r0 + row) * K + col;
      v = X[g];
      if (Agg) v += Agg[g];
    }
    As[idx] = (bf16)v;
  }
  // Stage B = W, swizzled: Bs[((nt*nkb+kkb)*32+lane)*16 + e]
  for (int idx = tid; idx < 8 * nkb * 512; idx += 128) {
    int e = idx & 15;
    int t = idx >> 4;
    int lane = t & 31; t >>= 5;
    int kkb = t % nkb, nt = t / nkb;
    int krow = kkb * 32 + e + (e & 8) + ((lane >> 4) << 3);
    int c = nt * 16 + (lane & 15);
    float v = (krow < K) ? W[krow * 128 + c] : 0.0f;
    Bs[idx] = (bf16)v;
  }
  __syncthreads();

  const int wave = tid >> 5;
  const int lane = tid & 31;
  const int lo = lane & 15;
  const int hi = lane >> 4;
  const int arow = wave * 16 + lo;

  v8f acc[8];
#pragma unroll
  for (int nt = 0; nt < 8; ++nt) acc[nt] = (v8f){0, 0, 0, 0, 0, 0, 0, 0};

  const v16bf* Asv = (const v16bf*)As;
  const v16bf* Bsv = (const v16bf*)Bs;
  for (int kkb = 0; kkb < nkb; ++kkb) {
    v16bf a = Asv[(arow * nkb + kkb) * 2 + hi];
#pragma unroll
    for (int nt = 0; nt < 8; ++nt) {
      v16bf b = Bsv[(nt * nkb + kkb) * 32 + lane];
      acc[nt] = __builtin_amdgcn_wmma_f32_16x16x32_bf16(
          false, a, false, b, (short)0, acc[nt], false, false);
    }
  }

  // Epilogue: C/D layout -> row = wave*16 + i + 8*hi, col = nt*16 + lo
#pragma unroll
  for (int nt = 0; nt < 8; ++nt) {
#pragma unroll
    for (int i = 0; i < 8; ++i) {
      int row = r0 + wave * 16 + i + hi * 8;
      int col = nt * 16 + lo;
      float v = acc[nt][i] + bias[col];
      if (applyRelu) v = fmaxf(v, 0.0f);
      Out[row * 128 + col] = v;
    }
  }
}

// ----------------------------------------------- per-node dot (logit head)
__global__ __launch_bounds__(256) void rowdot_kernel(
    const float* __restrict__ Hm, const float* __restrict__ w,
    const float* __restrict__ bptr, float* __restrict__ out, int n) {
  int gw = (blockIdx.x * 256 + threadIdx.x) >> 5;
  int lane = threadIdx.x & 31;
  if (gw >= n) return;
  const float* row = Hm + (size_t)gw * 128;
  float s = 0.0f;
  for (int j = lane; j < 128; j += 32) s += row[j] * w[j];
#pragma unroll
  for (int off = 16; off > 0; off >>= 1) s += __shfl_xor(s, off, 32);
  if (lane == 0) out[gw] = s + bptr[0];
}

// ---------------- segmented softmax + entropy + Gumbel-max sample (per graph)
__global__ __launch_bounds__(256) void seg_softmax_sample_kernel(
    const float* __restrict__ logits, float* __restrict__ o_loc,
    float* __restrict__ o_llp, float* __restrict__ o_lent,
    int* __restrict__ loc_idx, int S) {
  __shared__ float red[256];
  __shared__ int redi[256];
  const int b = blockIdx.x, tid = threadIdx.x;
  const int base = b * S;

  // max
  float tm = -3.4e38f;
  for (int n = tid; n < S; n += 256) tm = fmaxf(tm, logits[base + n]);
  red[tid] = tm; __syncthreads();
  for (int s = 128; s > 0; s >>= 1) {
    if (tid < s) red[tid] = fmaxf(red[tid], red[tid + s]);
    __syncthreads();
  }
  float m = red[0]; __syncthreads();

  // sum exp
  float ts = 0.0f;
  for (int n = tid; n < S; n += 256) ts += __expf(logits[base + n] - m);
  red[tid] = ts; __syncthreads();
  for (int s = 128; s > 0; s >>= 1) {
    if (tid < s) red[tid] += red[tid + s];
    __syncthreads();
  }
  float logZ = __logf(red[0]); __syncthreads();

  // entropy = -sum p*logp
  float te = 0.0f;
  for (int n = tid; n < S; n += 256) {
    float lp = (logits[base + n] - m) - logZ;
    te += __expf(lp) * lp;
  }
  red[tid] = te; __syncthreads();
  for (int s = 128; s > 0; s >>= 1) {
    if (tid < s) red[tid] += red[tid + s];
    __syncthreads();
  }
  float ent = -red[0]; __syncthreads();

  // Gumbel-max argmax (min-index tie-break)
  float bv = -3.4e38f; int bi = base;
  for (int n = tid; n < S; n += 256) {
    int g = base + n;
    float u = u01_hash(0x9E3779B9u ^ (unsigned)g);
    float pert = logits[g] - __logf(-__logf(u));
    if (pert > bv || (pert == bv && g < bi)) { bv = pert; bi = g; }
  }
  red[tid] = bv; redi[tid] = bi; __syncthreads();
  for (int s = 128; s > 0; s >>= 1) {
    if (tid < s) {
      float ov = red[tid + s]; int oi = redi[tid + s];
      if (ov > red[tid] || (ov == red[tid] && oi < redi[tid])) {
        red[tid] = ov; redi[tid] = oi;
      }
    }
    __syncthreads();
  }
  if (tid == 0) {
    int loc = redi[0];
    o_loc[b] = (float)loc;
    o_llp[b] = (logits[loc] - m) - logZ;
    o_lent[b] = ent;
    loc_idx[b] = loc;
  }
}

// ---------------------------------------------------------- global add pool
__global__ __launch_bounds__(128) void pool_kernel(
    const float* __restrict__ emb, float* __restrict__ pooled, int S) {
  int b = blockIdx.x, c = threadIdx.x;
  const float* p = emb + (size_t)b * S * 128 + c;
  float s = 0.0f;
  for (int i = 0; i < S; ++i) s += p[(size_t)i * 128];
  pooled[b * 128 + c] = s;
}

// -------------------- mutation actor/critic + location critic (B=512, tiny)
__global__ __launch_bounds__(128) void heads_kernel(
    const float* __restrict__ emb, const float* __restrict__ pooled,
    const int* __restrict__ loc_idx,
    const float* __restrict__ maW1, const float* __restrict__ mab1,
    const float* __restrict__ maW2, const float* __restrict__ mab2,
    const float* __restrict__ mcW1, const float* __restrict__ mcb1,
    const float* __restrict__ mcW2, const float* __restrict__ mcb2,
    const float* __restrict__ lcW1, const float* __restrict__ lcb1,
    const float* __restrict__ lcW2, const float* __restrict__ lcb2,
    float* __restrict__ o_mut, float* __restrict__ o_mlp,
    float* __restrict__ o_ment, float* __restrict__ o_lval,
    float* __restrict__ o_mval) {
  __shared__ float sel[128], pl[128], hid[128], red[128], lg[4];
  const int b = blockIdx.x, c = threadIdx.x;
  const int loc = loc_idx[b];
  sel[c] = emb[(size_t)loc * 128 + c];
  pl[c] = pooled[b * 128 + c];
  __syncthreads();

  // mutation actor: hid = relu(sel @ maW1 + mab1)
  float s = mab1[c];
  for (int j = 0; j < 128; ++j) s += sel[j] * maW1[j * 128 + c];
  hid[c] = fmaxf(s, 0.0f);
  __syncthreads();
  if (c < 4) {
    float t = mab2[c];
    for (int j = 0; j < 128; ++j) t += hid[j] * maW2[j * 4 + c];
    lg[c] = t;
  }
  __syncthreads();
  if (c == 0) {
    float m = fmaxf(fmaxf(lg[0], lg[1]), fmaxf(lg[2], lg[3]));
    float Z = 0.0f;
    for (int k = 0; k < 4; ++k) Z += __expf(lg[k] - m);
    float logZ = __logf(Z);
    float ent = 0.0f, bestg = -3.4e38f; int best = 0;
    for (int k = 0; k < 4; ++k) {
      float lp = lg[k] - m - logZ;
      ent -= __expf(lp) * lp;
      float u = u01_hash(0x7F4A7C15u ^ (unsigned)(b * 4 + k));
      float g = lg[k] - __logf(-__logf(u));
      if (g > bestg) { bestg = g; best = k; }
    }
    o_mut[b] = (float)best;
    o_mlp[b] = lg[best] - m - logZ;
    o_ment[b] = ent;
  }
  __syncthreads();

  // mutation critic on sel
  s = mcb1[c];
  for (int j = 0; j < 128; ++j) s += sel[j] * mcW1[j * 128 + c];
  red[c] = fmaxf(s, 0.0f) * mcW2[c];
  __syncthreads();
  for (int st = 64; st > 0; st >>= 1) {
    if (c < st) red[c] += red[c + st];
    __syncthreads();
  }
  if (c == 0) o_mval[b] = red[0] + mcb2[0];
  __syncthreads();

  // location critic on pooled
  s = lcb1[c];
  for (int j = 0; j < 128; ++j) s += pl[j] * lcW1[j * 128 + c];
  red[c] = fmaxf(s, 0.0f) * lcW2[c];
  __syncthreads();
  for (int st = 64; st > 0; st >>= 1) {
    if (c < st) red[c] += red[c + st];
    __syncthreads();
  }
  if (c == 0) o_lval[b] = red[0] + lcb2[0];
}

// ---------------------------------------------------------------- launcher
extern "C" void kernel_launch(void* const* d_in, const int* in_sizes, int n_in,
                              void* d_out, int out_size, void* d_ws, size_t ws_size,
                              hipStream_t stream) {
  const float* x   = (const float*)d_in[0];
  const int*   ei  = (const int*)d_in[1];
  const float* ea  = (const float*)d_in[2];
  const float* We0 = (const float*)d_in[5],  *be0 = (const float*)d_in[6];
  const float* W0  = (const float*)d_in[7],  *b0  = (const float*)d_in[8];
  const float* We1 = (const float*)d_in[9],  *be1 = (const float*)d_in[10];
  const float* W1  = (const float*)d_in[11], *b1  = (const float*)d_in[12];
  const float* We2 = (const float*)d_in[13], *be2 = (const float*)d_in[14];
  const float* W2  = (const float*)d_in[15], *b2  = (const float*)d_in[16];
  const float* laW1 = (const float*)d_in[17], *lab1 = (const float*)d_in[18];
  const float* laW2 = (const float*)d_in[19], *lab2 = (const float*)d_in[20];
  const float* maW1 = (const float*)d_in[21], *mab1 = (const float*)d_in[22];
  const float* maW2 = (const float*)d_in[23], *mab2 = (const float*)d_in[24];
  const float* lcW1 = (const float*)d_in[25], *lcb1 = (const float*)d_in[26];
  const float* lcW2 = (const float*)d_in[27], *lcb2 = (const float*)d_in[28];
  const float* mcW1 = (const float*)d_in[29], *mcb1 = (const float*)d_in[30];
  const float* mcW2 = (const float*)d_in[31], *mcb2 = (const float*)d_in[32];

  const int N = in_sizes[0] / 8;          // 262144
  const int E = in_sizes[2] / 4;          // 1048576
  const int B = in_sizes[4] - 1;          // 512
  const int S = N / B;                    // 512 nodes per graph (contiguous)
  const int* src = ei;
  const int* dst = ei + E;

  float* outf   = (float*)d_out;
  float* o_loc  = outf + 0 * B;
  float* o_mut  = outf + 1 * B;
  float* o_llp  = outf + 2 * B;
  float* o_mlp  = outf + 3 * B;
  float* o_lent = outf + 4 * B;
  float* o_ment = outf + 5 * B;
  float* o_lval = outf + 6 * B;
  float* o_mval = outf + 7 * B;
  float* emb    = outf + 8 * B;           // [N, 128]

  float* hbuf   = (float*)d_ws;                    // [N,128]
  float* aggbuf = hbuf + (size_t)N * 128;          // [N,128]
  // after the GNN, aggbuf is recycled: logits [N], pooled [B,128], loc_idx [B]
  float* logitsbuf = aggbuf;
  float* pooled    = aggbuf + N;
  int*   loc_idx   = (int*)(pooled + (size_t)B * 128);

  const int denseGrid = N / 64;

  // ---- layer 0 (d = 8)
  hipMemsetAsync(aggbuf, 0, (size_t)N * 8 * sizeof(float), stream);
  edge_msg8_kernel<<<(E * 8 + 255) / 256, 256, 0, stream>>>(x, src, dst, ea, We0, be0, aggbuf, E);
  dense_wmma_kernel<<<denseGrid, 128, 0, stream>>>(x, aggbuf, W0, b0, hbuf, 8, 1);

  // ---- layer 1
  hipMemsetAsync(aggbuf, 0, (size_t)N * 128 * sizeof(float), stream);
  edge_msg128_kernel<<<(E * 128 + 255) / 256, 256, 0, stream>>>(hbuf, src, dst, ea, We1, be1, aggbuf, E);
  dense_wmma_kernel<<<denseGrid, 128, 0, stream>>>(hbuf, aggbuf, W1, b1, hbuf, 128, 1);

  // ---- layer 2 -> emb (directly into d_out)
  hipMemsetAsync(aggbuf, 0, (size_t)N * 128 * sizeof(float), stream);
  edge_msg128_kernel<<<(E * 128 + 255) / 256, 256, 0, stream>>>(hbuf, src, dst, ea, We2, be2, aggbuf, E);
  dense_wmma_kernel<<<denseGrid, 128, 0, stream>>>(hbuf, aggbuf, W2, b2, emb, 128, 1);

  // ---- location actor: hidden = relu(emb @ laW1 + lab1), logits = hidden @ laW2 + lab2
  dense_wmma_kernel<<<denseGrid, 128, 0, stream>>>(emb, nullptr, laW1, lab1, hbuf, 128, 1);
  rowdot_kernel<<<N / 8, 256, 0, stream>>>(hbuf, laW2, lab2, logitsbuf, N);

  // ---- segmented softmax / entropy / Gumbel sampling
  seg_softmax_sample_kernel<<<B, 256, 0, stream>>>(logitsbuf, o_loc, o_llp, o_lent, loc_idx, S);

  // ---- pooling + small heads
  pool_kernel<<<B, 128, 0, stream>>>(emb, pooled, S);
  heads_kernel<<<B, 128, 0, stream>>>(emb, pooled, loc_idx,
                                      maW1, mab1, maW2, mab2,
                                      mcW1, mcb1, mcW2, mcb2,
                                      lcW1, lcb1, lcW2, lcb2,
                                      o_mut, o_mlp, o_ment, o_lval, o_mval);
}
